// LibragradGemma2Attention_24060406792846
// MI455X (gfx1250) — matrix operand order repaired and checked by reading, not compiled, not run on hardware
//
#include <hip/hip_runtime.h>
#include <hip/hip_bf16.h>

// Gemma2 attention for MI455X (gfx1250, wave32, WMMA).
// All GEMMs: C = A @ W^T with A,W row-major -> one WMMA kernel, 3 epilogue modes.
// f32 sources are converted to bf16 while staging tiles into LDS (v_perm pack);
// accumulate f32 in WMMA. Block tile 128Mx256N, 8 waves as 2x4, wave tile 64x64:
// 16 v_wmma per K-step against 16 ds_load_b128 fragment loads.

#define S_LEN   2048
#define HID     3584
#define NHEADS  16
#define NKVH    8
#define HDIM    256
#define SOFTCAP 50.0f
#define ATT_SCALE 0.0625f          // 256^-0.5
#define NEGINF  (-1000000000.0f)

typedef __attribute__((ext_vector_type(16))) __bf16 v16bf;
typedef __attribute__((ext_vector_type(8)))  float  v8f;

union Frag { v16bf bf; uint4 u4[2]; };

// Pack two f32 -> two bf16 in one dword: round-bias add + one v_perm_b32.
__device__ __forceinline__ unsigned int pack2bf(float lo, float hi) {
    unsigned int ulo = __float_as_uint(lo) + 0x8000u;   // round to nearest
    unsigned int uhi = __float_as_uint(hi) + 0x8000u;
    // bytes: {uhi[3],uhi[2],ulo[3],ulo[2]} = {bf16(hi), bf16(lo)}
    return __builtin_amdgcn_perm(uhi, ulo, 0x07060302u);
}

#define LDSTR 40   // 32 payload halves + 8 pad halves per row (bank stagger)

// Stage a 128x32 f32 tile -> bf16 LDS. 256 threads: thread t does row t/2,
// 16 consecutive K-elements at column (t&1)*16.
__device__ __forceinline__ void stage_half(const float* __restrict__ g, int ld,
                                           int rowBase, int k0,
                                           unsigned short* __restrict__ lds, int tid)
{
    const int row = tid >> 1;
    const int col = (tid & 1) << 4;
    const float4* src = reinterpret_cast<const float4*>(
        g + (size_t)(rowBase + row) * ld + k0 + col);
    union { unsigned int u[8]; uint4 u4[2]; } tmp;
#pragma unroll
    for (int i = 0; i < 4; ++i) {
        float4 v = src[i];
        tmp.u[i*2+0] = pack2bf(v.x, v.y);
        tmp.u[i*2+1] = pack2bf(v.z, v.w);
    }
    uint4* dst = reinterpret_cast<uint4*>(&lds[row * LDSTR + col]);
    dst[0] = tmp.u4[0];
    dst[1] = tmp.u4[1];
}

// Stage a 256x32 f32 tile -> bf16 LDS. 256 threads: thread t does full row t.
__device__ __forceinline__ void stage_row32(const float* __restrict__ g, int ld,
                                            int rowBase, int k0,
                                            unsigned short* __restrict__ lds, int tid)
{
    const float4* src = reinterpret_cast<const float4*>(
        g + (size_t)(rowBase + tid) * ld + k0);
    union { unsigned int u[16]; uint4 u4[4]; } tmp;
#pragma unroll
    for (int i = 0; i < 8; ++i) {
        float4 v = src[i];
        tmp.u[i*2+0] = pack2bf(v.x, v.y);
        tmp.u[i*2+1] = pack2bf(v.z, v.w);
    }
    uint4* dst = reinterpret_cast<uint4*>(&lds[tid * LDSTR]);
#pragma unroll
    for (int i = 0; i < 4; ++i) dst[i] = tmp.u4[i];
}

// mode 0: C[m*ldc+n] = v
// mode 1: C[n*ldcT+m] = v                       (transposed store, for V^T)
// mode 2: scale -> tanh softcap -> causal mask  (scores into attn buffer)
__global__ __launch_bounds__(256)
void gemm_bf16_wmma(const float* __restrict__ A, const float* __restrict__ W,
                    float* __restrict__ C,
                    int K, int lda, int ldw, int ldc, int ldcT,
                    unsigned long long aStride, unsigned long long wStride,
                    unsigned long long cStride, int gqaShift, int mode)
{
    __shared__ __align__(16) unsigned short sA[128 * LDSTR];
    __shared__ __align__(16) unsigned short sB[256 * LDSTR];

    const int tid  = threadIdx.x;
    const int lane = tid & 31;
    const int wave = tid >> 5;
    const int wm   = wave & 1;     // 2 waves tile M (64 rows each)
    const int wn   = wave >> 1;    // 4 waves tile N (64 cols each)
    const int grp  = lane >> 4;    // half-wave group
    const int ln   = lane & 15;

    const int nBase = blockIdx.x * 256;
    const int mBase = blockIdx.y * 128;
    const int z     = blockIdx.z;

    A += (unsigned long long)z * aStride;
    W += (unsigned long long)(z >> gqaShift) * wStride;   // GQA: kv head = h>>1
    C += (unsigned long long)z * cStride;

    v8f acc[4][4];
#pragma unroll
    for (int i = 0; i < 4; ++i)
#pragma unroll
        for (int j = 0; j < 4; ++j)
#pragma unroll
            for (int r = 0; r < 8; ++r) acc[i][j][r] = 0.0f;

    for (int k0 = 0; k0 < K; k0 += 32) {
        stage_half (A, lda, mBase, k0, sA, tid);
        stage_row32(W, ldw, nBase, k0, sB, tid);
        if (k0 + 32 < K) {   // pull next K-tile toward L2 (global_prefetch_b8)
            __builtin_prefetch(A + (size_t)(mBase + (tid >> 1)) * lda + k0 + 32 + ((tid & 1) << 4), 0, 0);
            __builtin_prefetch(W + (size_t)(nBase + tid) * ldw + k0 + 32, 0, 0);
        }
        __syncthreads();

        // Fragment loads per ISA 16-bit A/B layouts (two b128 LDS reads each).
        Frag a[4], b[4];
#pragma unroll
        for (int i = 0; i < 4; ++i) {
            const int r   = wm * 64 + i * 16 + ln;
            const int off = r * LDSTR + grp * 8;           // K = grp*8 .. +7
            a[i].u4[0] = *reinterpret_cast<const uint4*>(&sA[off]);
            a[i].u4[1] = *reinterpret_cast<const uint4*>(&sA[off + 16]); // K = 16+grp*8 ..
        }
#pragma unroll
        for (int j = 0; j < 4; ++j) {
            const int r   = wn * 64 + j * 16 + ln;
            const int off = r * LDSTR + grp * 8;           // K = grp*8 .. +7, 16+grp*8 .. +7
            b[j].u4[0] = *reinterpret_cast<const uint4*>(&sB[off]);
            b[j].u4[1] = *reinterpret_cast<const uint4*>(&sB[off + 16]);
        }
#pragma unroll
        for (int i = 0; i < 4; ++i)
#pragma unroll
            for (int j = 0; j < 4; ++j)
                acc[i][j] = __builtin_amdgcn_wmma_f32_16x16x32_bf16(
                    false, a[i].bf, false, b[j].bf, (short)0, acc[i][j], false, false);
        __syncthreads();
    }

    // Epilogue: C VGPR r holds M = r + grp*8, N = ln (per ISA C/D layout).
#pragma unroll
    for (int i = 0; i < 4; ++i) {
#pragma unroll
        for (int j = 0; j < 4; ++j) {
#pragma unroll
            for (int r = 0; r < 8; ++r) {
                const int m = mBase + wm * 64 + i * 16 + grp * 8 + r;
                const int n = nBase + wn * 64 + j * 16 + ln;
                float v = acc[i][j][r];
                if (mode == 2) {
                    v *= ATT_SCALE;
                    v = tanhf(v * (1.0f / SOFTCAP)) * SOFTCAP;
                    if (n > m) v += NEGINF;    // causal mask (m=query, n=key)
                }
                if (mode == 1) C[(size_t)n * ldcT + m] = v;
                else           C[(size_t)m * ldc  + n] = v;
            }
        }
    }
}

// RoPE in place: x is (S, nheads, 256); cos/sin are (S, 256) with halves duplicated.
__global__ void rope_kernel(float* __restrict__ x, const float* __restrict__ cosT,
                            const float* __restrict__ sinT, int nheads)
{
    const int idx = blockIdx.x * blockDim.x + threadIdx.x;
    const int d = idx & 127;
    const int h = (idx >> 7) % nheads;
    const int s = idx / (128 * nheads);
    float* p = x + (size_t)s * nheads * HDIM + h * HDIM;
    const float c  = cosT[s * HDIM + d];
    const float sn = sinT[s * HDIM + d];
    const float a = p[d], b = p[d + 128];
    p[d]       = a * c - b * sn;   // x1*cos - x2*sin
    p[d + 128] = b * c + a * sn;   // x2*cos + x1*sin
}

// Row softmax in place over 2048-wide rows; one 256-thread block per row.
__global__ __launch_bounds__(256)
void softmax_rows(float* __restrict__ attn)
{
    __shared__ float red[8];
    float* p = attn + (size_t)blockIdx.x * S_LEN;
    const int tid = threadIdx.x;

    float vals[8];
    float m = -3.0e38f;
#pragma unroll
    for (int i = 0; i < 8; ++i) { vals[i] = p[tid + i * 256]; m = fmaxf(m, vals[i]); }
#pragma unroll
    for (int off = 16; off; off >>= 1) m = fmaxf(m, __shfl_xor(m, off, 32));
    if ((tid & 31) == 0) red[tid >> 5] = m;
    __syncthreads();
#pragma unroll
    for (int w = 0; w < 8; ++w) m = fmaxf(m, red[w]);

    float sum = 0.0f;
#pragma unroll
    for (int i = 0; i < 8; ++i) { vals[i] = expf(vals[i] - m); sum += vals[i]; }
#pragma unroll
    for (int off = 16; off; off >>= 1) sum += __shfl_xor(sum, off, 32);
    __syncthreads();
    if ((tid & 31) == 0) red[tid >> 5] = sum;
    __syncthreads();
    sum = 0.0f;
#pragma unroll
    for (int w = 0; w < 8; ++w) sum += red[w];

    const float inv = 1.0f / sum;
#pragma unroll
    for (int i = 0; i < 8; ++i) p[tid + i * 256] = vals[i] * inv;
}

extern "C" void kernel_launch(void* const* d_in, const int* in_sizes, int n_in,
                              void* d_out, int out_size, void* d_ws, size_t ws_size,
                              hipStream_t stream)
{
    (void)in_sizes; (void)n_in; (void)out_size; (void)ws_size;
    const float* hs   = (const float*)d_in[0];
    const float* cosT = (const float*)d_in[1];
    const float* sinT = (const float*)d_in[2];
    // d_in[3] = attention_mask (causal mask recomputed in epilogue)
    const float* wq   = (const float*)d_in[4];
    const float* wk   = (const float*)d_in[5];
    const float* wv   = (const float*)d_in[6];
    const float* wo   = (const float*)d_in[7];

    float* outp = (float*)d_out;                        // (2048, 3584)
    float* attn = outp + (size_t)S_LEN * HID;           // (16, 2048, 2048)

    // Workspace (f32): q 33.5MB | k 16.8MB | v^T 16.8MB | ctx 33.5MB  (~96MB)
    float* q    = (float*)d_ws;                         // (2048, 4096)
    float* kbuf = q    + (size_t)S_LEN * (NHEADS * HDIM);
    float* vT   = kbuf + (size_t)S_LEN * (NKVH * HDIM); // (2048 d, 2048 s)
    float* ctx  = vT   + (size_t)S_LEN * (NKVH * HDIM); // (2048, 4096)

    dim3 blk(256);

    // Q = hs @ wq^T   (2048x3584 @ 3584x4096)
    gemm_bf16_wmma<<<dim3((NHEADS*HDIM)/256, S_LEN/128, 1), blk, 0, stream>>>(
        hs, wq, q, HID, HID, HID, NHEADS*HDIM, 0, 0ULL, 0ULL, 0ULL, 0, 0);
    // K = hs @ wk^T
    gemm_bf16_wmma<<<dim3((NKVH*HDIM)/256, S_LEN/128, 1), blk, 0, stream>>>(
        hs, wk, kbuf, HID, HID, HID, NKVH*HDIM, 0, 0ULL, 0ULL, 0ULL, 0, 0);
    // V^T = (hs @ wv^T)^T  -> stored (d_total, s) so attn@V sees contiguous K
    gemm_bf16_wmma<<<dim3((NKVH*HDIM)/256, S_LEN/128, 1), blk, 0, stream>>>(
        hs, wv, vT, HID, HID, HID, 0, S_LEN, 0ULL, 0ULL, 0ULL, 0, 1);

    // RoPE on Q and K (f32, in place)
    rope_kernel<<<(S_LEN*NHEADS*(HDIM/2))/256, blk, 0, stream>>>(q, cosT, sinT, NHEADS);
    rope_kernel<<<(S_LEN*NKVH  *(HDIM/2))/256, blk, 0, stream>>>(kbuf, cosT, sinT, NKVH);

    // scores[h] = softcap(q_h @ k_{h>>1}^T * scale) + causal  -> attn buffer
    gemm_bf16_wmma<<<dim3(S_LEN/256, S_LEN/128, NHEADS), blk, 0, stream>>>(
        q, kbuf, attn, HDIM, NHEADS*HDIM, NKVH*HDIM, S_LEN, 0,
        (unsigned long long)HDIM, (unsigned long long)HDIM,
        (unsigned long long)S_LEN * S_LEN, 1, 2);

    // softmax rows in place (this is the returned attn output)
    softmax_rows<<<NHEADS * S_LEN, blk, 0, stream>>>(attn);

    // ctx[:, h*256:(h+1)*256] = attn_h @ v_{h>>1}
    gemm_bf16_wmma<<<dim3(HDIM/256, S_LEN/128, NHEADS), blk, 0, stream>>>(
        attn, vT, ctx, S_LEN, S_LEN, S_LEN, NHEADS*HDIM, 0,
        (unsigned long long)S_LEN * S_LEN, (unsigned long long)HDIM * S_LEN,
        (unsigned long long)HDIM, 1, 0);

    // attn_output = ctx @ wo^T  (2048x4096 @ 4096x3584)
    gemm_bf16_wmma<<<dim3(HID/256, S_LEN/128, 1), blk, 0, stream>>>(
        ctx, wo, outp, NHEADS*HDIM, NHEADS*HDIM, NHEADS*HDIM, HID, 0,
        0ULL, 0ULL, 0ULL, 0, 0);
}
